// WindowAttention_12498354831833
// MI455X (gfx1250) — matrix operand order repaired and checked by reading, not compiled, not run on hardware
//
#include <hip/hip_runtime.h>
#include <hip/hip_bf16.h>
#include <math.h>

// ---------------------------------------------------------------------------
// Window "mutual" attention, fused per-window, f16 WMMA compute / f32 accum.
// One workgroup (256 threads = 8 wave32) per window (1024 windows).
// All intermediates (X, Q, K, V^T, P, O) live in LDS (~306KB of the 320KB/WGP).
// Wave-uniform values are scalarized via readfirstlane so all phase control
// flow is SALU and EXEC is provably all-ones around every v_wmma.
// ---------------------------------------------------------------------------

typedef _Float16 v16h __attribute__((ext_vector_type(16)));
typedef _Float16 v8h  __attribute__((ext_vector_type(8)));
typedef _Float16 v4h  __attribute__((ext_vector_type(4)));
typedef float    v8f  __attribute__((ext_vector_type(8)));
typedef float    v4f  __attribute__((ext_vector_type(4)));

#define DIMC   192
#define HEADS  6
#define HDIM   32
#define NTOK   128
#define NWIN   64
#define QK_SCALE 0.17677669529663687f   // 32^-0.5

// LDS row strides (in halves); all row starts 16-byte aligned
#define SX  200   // X buffer (also reused as P with stride SP)
#define SQK 200   // Q and K buffers
#define SVT 136   // V^T buffer  [dim][token]
#define SP  136   // P (softmaxed scores), aliased into sX region
#define SO  392   // O buffer    [token][384]

#define SMEM_HALVES (NTOK*SX + NTOK*SQK + NTOK*SQK + DIMC*SVT + NTOK*SO)

__device__ __forceinline__ v16h cat8(v8h lo, v8h hi) {
  v16h r;
#pragma unroll
  for (int i = 0; i < 8; ++i) { r[i] = lo[i]; r[i + 8] = hi[i]; }
  return r;
}

// A-matrix fragment (16x32 f16), per CDNA5 ISA layout:
// lanes 0-15: M=lane, K=[k0..k0+7] in v0-3, K=[k0+16..k0+23] in v4-7
// lanes 16-31: M=lane-16, K=[k0+8..k0+15] / [k0+24..k0+31]
__device__ __forceinline__ v16h load_a(const _Float16* buf, int row0, int stride, int k0) {
  const int lane = threadIdx.x & 31;
  const int m = lane & 15, hh = lane >> 4;
  const _Float16* p = buf + (row0 + m) * stride + k0 + hh * 8;
  v8h lo = *(const v8h*)p;
  v8h hi = *(const v8h*)(p + 16);
  return cat8(lo, hi);
}

// B-matrix fragment (32x16 f16) from a [N][K]-major ("B^T") buffer:
// lanes 0-15: N=lane, K=[k0..k0+7],[k0+16..k0+23]; lanes 16-31: N=lane-16, K=[k0+8..k0+15],[k0+24..k0+31]
__device__ __forceinline__ v16h load_b(const _Float16* bt, int col0, int stride, int k0) {
  const int lane = threadIdx.x & 31;
  const int n = lane & 15, hh = lane >> 4;
  const _Float16* p = bt + (col0 + n) * stride + k0 + hh * 8;
  v8h lo = *(const v8h*)p;
  v8h hi = *(const v8h*)(p + 16);
  return cat8(lo, hi);
}

__device__ __forceinline__ v8f wmma_f16(v16h a, v16h b, v8f c) {
  // D = A*B + C, f32 accumulate
  return __builtin_amdgcn_wmma_f32_16x16x32_f16(false, a, false, b, (short)0, c, false, false);
}

// D tile (16x16 f32, C-layout: VGPR r -> M=r (lanes 0-15) / M=r+8 (lanes 16-31))
__device__ __forceinline__ void store_d(_Float16* buf, int row0, int stride, int col0,
                                        v8f d, float scale) {
  const int lane = threadIdx.x & 31;
  const int n = lane & 15, hh = lane >> 4;
#pragma unroll
  for (int r = 0; r < 8; ++r)
    buf[(row0 + r + hh * 8) * stride + col0 + n] = (_Float16)(d[r] * scale);
}

// transposed store: D(m,n) -> buf[n][m]  (used to build V^T; contiguous -> b128)
__device__ __forceinline__ void store_dt(_Float16* buf, int row0, int stride, int col0, v8f d) {
  const int lane = threadIdx.x & 31;
  const int n = lane & 15, hh = lane >> 4;
  v8h h;
#pragma unroll
  for (int r = 0; r < 8; ++r) h[r] = (_Float16)d[r];
  *(v8h*)(buf + (col0 + n) * stride + row0 + hh * 8) = h;
}

// ---------------------------------------------------------------------------
// Prep kernel: pack weights transposed to f16, build sine-pos table and the
// relative-position bias tensor (REL_IDX evaluated analytically).
// ---------------------------------------------------------------------------
__global__ void wattn_prep(const float* __restrict__ wself, const float* __restrict__ wmut,
                           const float* __restrict__ wproj, const float* __restrict__ table,
                           _Float16* __restrict__ wselfT, _Float16* __restrict__ wmutT,
                           _Float16* __restrict__ wprojT, float* __restrict__ pos,
                           float* __restrict__ bias6) {
  const int NT1 = 576 * 192;     // qkv weights (transposed elems)
  const int NT2 = 192 * 384;     // proj weights (transposed elems)
  const int NT3 = 128 * 192;     // positional encoding
  const int NT4 = 6 * 128 * 128; // relative bias
  const int total = NT1 + NT2 + NT3 + NT4;
  for (int i = blockIdx.x * blockDim.x + threadIdx.x; i < total;
       i += gridDim.x * blockDim.x) {
    if (i < NT1) {
      int o = i / 192, c = i % 192;            // wT[o][c] = w[c][o]
      wselfT[i] = (_Float16)wself[c * 576 + o];
      wmutT[i]  = (_Float16)wmut[c * 576 + o];
    } else if (i < NT1 + NT2) {
      int j = i - NT1;
      int o = j / 384, c = j % 384;            // wprojT[o][c] = wproj[c][o]
      wprojT[j] = (_Float16)wproj[c * 192 + o];
    } else if (i < NT1 + NT2 + NT3) {
      int j = i - NT1 - NT2;
      int n = j / 192, c = j % 192;
      int t = n & 63;                          // pos tiled (2,1): row n -> n%64
      int yi = t >> 3, xi = t & 7;
      const float twopi = 6.283185307179586f;
      float val; int cc;
      if (c < 96) { val = (float)(yi + 1) / (8.0f + 1e-6f) * twopi; cc = c; }
      else        { val = (float)(xi + 1) / (8.0f + 1e-6f) * twopi; cc = c - 96; }
      int pm = cc >> 1;
      float dt = powf(10000.0f, (float)(2 * pm) / 96.0f);
      float a = val / dt;
      pos[j] = (cc & 1) ? cosf(a) : sinf(a);
    } else {
      int j = i - NT1 - NT2 - NT3;
      int h = j / 16384;
      int m = (j >> 7) & 127;
      int n = j & 127;
      int dm = m >> 6, hm = (m >> 3) & 7, wm = m & 7;
      int dn = n >> 6, hn = (n >> 3) & 7, wn = n & 7;
      int idx = (dm - dn + 1) * 225 + (hm - hn + 7) * 15 + (wm - wn + 7);
      bias6[j] = table[idx * 6 + h];           // bias6[h][m][n]
    }
  }
}

// ---------------------------------------------------------------------------
// Main fused kernel
// ---------------------------------------------------------------------------
__global__ __launch_bounds__(256, 1)
void wattn_main(const float* __restrict__ x, const float* __restrict__ mask,
                const _Float16* __restrict__ wselfT, const _Float16* __restrict__ wmutT,
                const _Float16* __restrict__ wprojT,
                const float* __restrict__ pos, const float* __restrict__ bias6,
                const float* __restrict__ bproj, float* __restrict__ out) {
  extern __shared__ __align__(16) _Float16 smem[];
  _Float16* sX  = smem;                 // 128*SX   (reused as P, stride SP)
  _Float16* sQ  = sX  + NTOK * SX;      // 128*SQK  (Q, pre-scaled)
  _Float16* sK  = sQ  + NTOK * SQK;     // 128*SQK
  _Float16* sVT = sK  + NTOK * SQK;     // 192*SVT  (V^T: [dim][token])
  _Float16* sO  = sVT + DIMC * SVT;     // 128*SO   (cols 0..191 mut, 192..383 self)

  const int b    = blockIdx.x;
  const int tid  = threadIdx.x;
  // wave id is uniform within a wave: force it into an SGPR so every derived
  // loop bound / branch is SALU and EXEC stays all-ones around the WMMAs.
  const int wv   = __builtin_amdgcn_readfirstlane(tid >> 5);   // 0..7
  const int lane = tid & 31;
  const int ln   = lane & 15;
  const int lh   = lane >> 4;
  const int m0   = wv * 16;             // this wave's M strip
  const int widx = b & (NWIN - 1);
  const float* xb   = x    + (size_t)b * NTOK * DIMC;
  const float* mwin = mask + (size_t)widx * NTOK * NTOK;

  // ---- phase 1: X -> LDS f16 (vectorized: float4 load -> 4-half store)
  for (int i = tid; i < NTOK * DIMC / 4; i += 256) {
    int r = i / (DIMC / 4), c4 = (i % (DIMC / 4)) * 4;
    v4f v = *(const v4f*)(xb + r * DIMC + c4);
    v4h h; h[0] = (_Float16)v[0]; h[1] = (_Float16)v[1];
           h[2] = (_Float16)v[2]; h[3] = (_Float16)v[3];
    *(v4h*)(sX + r * SX + c4) = h;
  }
  __syncthreads();

  // ---- phase 2: QKV(self) = X @ Wself ; waves split N-tiles (weights hit L2 once)
  for (int nt = wv; nt < 36; nt += 8) {
    int o0 = nt * 16;
    v16h bf[6];
#pragma unroll
    for (int kt = 0; kt < 6; ++kt) bf[kt] = load_b(wselfT, o0, DIMC, kt * 32);
    for (int mt = 0; mt < 8; ++mt) {
      v8f acc = {};
#pragma unroll
      for (int kt = 0; kt < 6; ++kt)
        acc = wmma_f16(load_a(sX, mt * 16, SX, kt * 32), bf[kt], acc);
      if (o0 < DIMC)            store_d(sQ, mt * 16, SQK, o0, acc, QK_SCALE);
      else if (o0 < 2 * DIMC)   store_d(sK, mt * 16, SQK, o0 - DIMC, acc, 1.0f);
      else                      store_dt(sVT, mt * 16, SVT, o0 - 2 * DIMC, acc);
    }
  }
  __syncthreads();

  // ---- phase 3: self attention (bias + mask), writes sO cols [192,384)
  for (int h = 0; h < HEADS; ++h) {
    v16h aq = load_a(sQ, m0, SQK, h * HDIM);
    const float* bp = bias6 + h * NTOK * NTOK;
    v8f s[8];
#pragma unroll
    for (int nt = 0; nt < 8; ++nt) {
      v8f c = {};
      c = wmma_f16(aq, load_b(sK, nt * 16, SQK, h * HDIM), c);  // K=32=HDIM: 1 WMMA
#pragma unroll
      for (int r = 0; r < 8; ++r) {
        int m = m0 + r + lh * 8;
        int n = nt * 16 + ln;
        c[r] += bp[m * NTOK + n] + mwin[m * NTOK + n];
      }
      s[nt] = c;
    }
    // row softmax over 128 cols (C-layout rows live in (r, half) pairs)
#pragma unroll
    for (int r = 0; r < 8; ++r) {
      float mx = -3.0e38f;
#pragma unroll
      for (int nt = 0; nt < 8; ++nt) mx = fmaxf(mx, s[nt][r]);
#pragma unroll
      for (int off = 1; off < 16; off <<= 1) mx = fmaxf(mx, __shfl_xor(mx, off, 32));
      float sum = 0.0f;
#pragma unroll
      for (int nt = 0; nt < 8; ++nt) { float e = __expf(s[nt][r] - mx); s[nt][r] = e; sum += e; }
#pragma unroll
      for (int off = 1; off < 16; off <<= 1) sum += __shfl_xor(sum, off, 32);
      float inv = 1.0f / sum;
      int m = m0 + r + lh * 8;
#pragma unroll
      for (int nt = 0; nt < 8; ++nt)
        sX[m * SP + nt * 16 + ln] = (_Float16)(s[nt][r] * inv);  // P, wave-private rows
    }
    // P @ V
#pragma unroll
    for (int dt = 0; dt < 2; ++dt) {
      v8f o = {};
#pragma unroll
      for (int kt = 0; kt < 4; ++kt)
        o = wmma_f16(load_a(sX, m0, SP, kt * 32),
                     load_b(sVT, h * HDIM + dt * 16, SVT, kt * 32), o);
      store_d(sO, m0, SO, DIMC + h * HDIM + dt * 16, o, 1.0f);
    }
  }
  __syncthreads();

  // ---- phase 4: X+pos -> LDS f16 (overwrites X/P region)
  for (int i = tid; i < NTOK * DIMC / 4; i += 256) {
    int r = i / (DIMC / 4), c4 = (i % (DIMC / 4)) * 4;
    v4f v = *(const v4f*)(xb + r * DIMC + c4);
    v4f p = *(const v4f*)(pos + r * DIMC + c4);
    v4h h; h[0] = (_Float16)(v[0] + p[0]); h[1] = (_Float16)(v[1] + p[1]);
           h[2] = (_Float16)(v[2] + p[2]); h[3] = (_Float16)(v[3] + p[3]);
    *(v4h*)(sX + r * SX + c4) = h;
  }
  __syncthreads();

  // ---- phase 5: QKV(mut) = (X+pos) @ Wmut
  for (int nt = wv; nt < 36; nt += 8) {
    int o0 = nt * 16;
    v16h bf[6];
#pragma unroll
    for (int kt = 0; kt < 6; ++kt) bf[kt] = load_b(wmutT, o0, DIMC, kt * 32);
    for (int mt = 0; mt < 8; ++mt) {
      v8f acc = {};
#pragma unroll
      for (int kt = 0; kt < 6; ++kt)
        acc = wmma_f16(load_a(sX, mt * 16, SX, kt * 32), bf[kt], acc);
      if (o0 < DIMC)            store_d(sQ, mt * 16, SQK, o0, acc, QK_SCALE);
      else if (o0 < 2 * DIMC)   store_d(sK, mt * 16, SQK, o0 - DIMC, acc, 1.0f);
      else                      store_dt(sVT, mt * 16, SVT, o0 - 2 * DIMC, acc);
    }
  }
  __syncthreads();

  // ---- phase 6: mutual attention (halves swapped), writes sO cols [0,192)
  {
    const int qb   = (wv < 4) ? 64 + m0 : m0 - 64;  // x1: q2,k1,v1 ; x2: q1,k2,v2
    const int kvb  = (wv < 4) ? 0 : 64;
    const int mr0  = (wv < 4) ? m0 : m0 - 64;       // mask row base (mask[:, :64, :64])
    for (int h = 0; h < HEADS; ++h) {
      v16h aq = load_a(sQ, qb, SQK, h * HDIM);
      v8f s[4];
#pragma unroll
      for (int nt = 0; nt < 4; ++nt) {
        v8f c = {};
        c = wmma_f16(aq, load_b(sK, kvb + nt * 16, SQK, h * HDIM), c);
#pragma unroll
        for (int r = 0; r < 8; ++r)
          c[r] += mwin[(mr0 + r + lh * 8) * NTOK + nt * 16 + ln];
        s[nt] = c;
      }
#pragma unroll
      for (int r = 0; r < 8; ++r) {
        float mx = -3.0e38f;
#pragma unroll
        for (int nt = 0; nt < 4; ++nt) mx = fmaxf(mx, s[nt][r]);
#pragma unroll
        for (int off = 1; off < 16; off <<= 1) mx = fmaxf(mx, __shfl_xor(mx, off, 32));
        float sum = 0.0f;
#pragma unroll
        for (int nt = 0; nt < 4; ++nt) { float e = __expf(s[nt][r] - mx); s[nt][r] = e; sum += e; }
#pragma unroll
        for (int off = 1; off < 16; off <<= 1) sum += __shfl_xor(sum, off, 32);
        float inv = 1.0f / sum;
        int m = m0 + r + lh * 8;
#pragma unroll
        for (int nt = 0; nt < 4; ++nt)
          sX[m * SP + nt * 16 + ln] = (_Float16)(s[nt][r] * inv);
      }
#pragma unroll
      for (int dt = 0; dt < 2; ++dt) {
        v8f o = {};
#pragma unroll
        for (int kt = 0; kt < 2; ++kt)
          o = wmma_f16(load_a(sX, m0, SP, kt * 32),
                       load_b(sVT, h * HDIM + dt * 16, SVT, kvb + kt * 32), o);
        store_d(sO, m0, SO, h * HDIM + dt * 16, o, 1.0f);
      }
    }
  }
  __syncthreads();

  // ---- phase 7: Y = O(128x384) @ Wproj + b_proj  -> global f32
  float* op = out + (size_t)b * NTOK * DIMC;
  for (int nt = wv; nt < 12; nt += 8) {
    int o0 = nt * 16;
    v16h bf[12];
#pragma unroll
    for (int kt = 0; kt < 12; ++kt) bf[kt] = load_b(wprojT, o0, 2 * DIMC, kt * 32);
    float bb = bproj[o0 + ln];
    for (int mt = 0; mt < 8; ++mt) {
      v8f acc = {};
#pragma unroll
      for (int kt = 0; kt < 12; ++kt)
        acc = wmma_f16(load_a(sO, mt * 16, SO, kt * 32), bf[kt], acc);
#pragma unroll
      for (int r = 0; r < 8; ++r)
        op[(mt * 16 + r + lh * 8) * DIMC + o0 + ln] = acc[r] + bb;
    }
  }
}

// ---------------------------------------------------------------------------
extern "C" void kernel_launch(void* const* d_in, const int* in_sizes, int n_in,
                              void* d_out, int out_size, void* d_ws, size_t ws_size,
                              hipStream_t stream) {
  (void)in_sizes; (void)n_in; (void)out_size; (void)ws_size;
  const float* x     = (const float*)d_in[0];
  const float* mask  = (const float*)d_in[1];
  const float* table = (const float*)d_in[2];
  const float* wself = (const float*)d_in[3];
  const float* wmut  = (const float*)d_in[4];
  const float* wproj = (const float*)d_in[5];
  const float* bproj = (const float*)d_in[6];
  float* out = (float*)d_out;

  // workspace layout (bytes)
  char* ws = (char*)d_ws;
  _Float16* wselfT = (_Float16*)(ws);             // 576*192 f16 = 221184 B
  _Float16* wmutT  = (_Float16*)(ws + 221184);    // 221184 B
  _Float16* wprojT = (_Float16*)(ws + 442368);    // 192*384 f16 = 147456 B
  float*    pos    = (float*)(ws + 589824);       // 128*192 f32 = 98304 B
  float*    bias6  = (float*)(ws + 688128);       // 6*128*128 f32 = 393216 B

  wattn_prep<<<300, 256, 0, stream>>>(wself, wmut, wproj, table,
                                      wselfT, wmutT, wprojT, pos, bias6);

  const size_t smem_bytes = (size_t)SMEM_HALVES * sizeof(_Float16);  // 306176 B
  wattn_main<<<1024, 256, smem_bytes, stream>>>(x, mask, wselfT, wmutT, wprojT,
                                                pos, bias6, bproj, out);
}
